// ConsolidationNetwork_9560597201158
// MI455X (gfx1250) — matrix operand order repaired and checked by reading, not compiled, not run on hardware
//
#include <hip/hip_runtime.h>
#include <stdint.h>

#define N_ 1024
#define G_ 128
#define T_ 500
#define B_ 128
#define P_ 10
#define A_COEF (0.05f / 0.15f)                 // DT/TAU
#define NSCALE 0.36742346141747673f            // sqrt(2*0.15^2*(TAU/DT))

typedef __attribute__((ext_vector_type(16))) __bf16 v16bf;
typedef __attribute__((ext_vector_type(8)))  float  v8f;

__device__ __forceinline__ uint32_t lds_offset(const void* p) {
  return (uint32_t)(uintptr_t)(__attribute__((address_space(3))) const void*)p;
}

// CDNA5 async copy global -> LDS (tracked by ASYNCcnt). VDST = LDS byte addr.
__device__ __forceinline__ void async_b128(const void* g, uint32_t loff) {
  asm volatile("global_load_async_to_lds_b128 %0, %1, off"
               :: "v"(loff), "v"(g) : "memory");
}
__device__ __forceinline__ void wait_async0() {
  asm volatile("s_wait_asynccnt 0" ::: "memory");
}

__device__ __forceinline__ float softplus_f(float x) {
  // stable softplus: max(x,0) + log1p(exp(-|x|))
  return fmaxf(x, 0.0f) + __logf(1.0f + __expf(-fabsf(x)));
}

// ---------------------------------------------------------------------------
// Kernel 1: J_eff = J + (U * B_bg) @ V, stored bf16 pre-swizzled into the
// WMMA A-operand layout (ISA 7.12.2, 16-bit A 16x32 table):
//   lanes 0-15 : M=0..15, dwords 0-3 hold K pairs 0..7, dwords 4-7 -> K 16..23
//   lanes 16-31: M=0..15, dwords 0-3 hold K pairs 8..15, dwords 4-7 -> K 24..31
// Block (tile_m, tile_k) = 1024 bytes; lane fragment = 32 contiguous bytes.
// ---------------------------------------------------------------------------
__global__ void jpack_kernel(const float* __restrict__ J, const float* __restrict__ U,
                             const float* __restrict__ V, const float* __restrict__ Bbg,
                             __bf16* __restrict__ Jp) {
  int idx = blockIdx.x * 256 + threadIdx.x;     // 0 .. N*N-1
  int m = idx >> 10, k = idx & 1023;
  float acc = J[idx];
  #pragma unroll 4
  for (int g = 0; g < G_; ++g)
    acc = fmaf(U[m * G_ + g] * Bbg[g], V[g * N_ + k], acc);

  int tile_m = m >> 4, m_in = m & 15, tk = k >> 5, kin = k & 31;
  int grp, dw;
  if      (kin < 8)  { grp = 0; dw =      kin        >> 1; }
  else if (kin < 16) { grp = 1; dw =     (kin - 8)   >> 1; }
  else if (kin < 24) { grp = 0; dw = 4 + ((kin - 16) >> 1); }
  else               { grp = 1; dw = 4 + ((kin - 24) >> 1); }
  int lane = m_in + 16 * grp;
  size_t off = ((size_t)(tile_m * 32 + tk) * 32 + lane) * 16 + dw * 2 + (kin & 1);
  Jp[off] = (__bf16)acc;
}

// ---------------------------------------------------------------------------
// Kernel 2: persistent per-column-block recurrence.
// grid = 8 blocks (16 batch columns each), 1024 threads = 32 waves.
// Wave w owns output row tiles 2w and 2w+1 (32 rows of x).
// ---------------------------------------------------------------------------
__global__ void __launch_bounds__(1024, 1)
net_kernel(const float* __restrict__ targets, const float* __restrict__ pulses,
           const float* __restrict__ B_m1,    const float* __restrict__ Wout,
           const float* __restrict__ I_go,    const float* __restrict__ x0,
           const float* __restrict__ noise,   const int*   __restrict__ triggers,
           const __bf16* __restrict__ Jp,     float* __restrict__ out) {
  // rm1 image, bf16 pre-swizzled WMMA B fragments (lane<->N, K table order)  : 32 KB
  __shared__ __bf16 ldsB[N_ * 16];
  // noise tile for current step, f32 [m][n]                                 : 64 KB
  __shared__ float  ldsN[N_ * 16];
  __shared__ float  pos_buf[16];

  const int tid  = threadIdx.x;
  const int wave = tid >> 5;
  const int lane = tid & 31;
  const int half = lane >> 4;       // C/D layout: lanes 16-31 hold M = 8..15
  const int nloc = lane & 15;       // column within block
  const int col0 = blockIdx.x * 16;
  const int ng   = col0 + nloc;
  const int rt0  = wave * 2, rt1 = wave * 2 + 1;

  float x[2][8], bm[2][8], ig[2][8], wo[2][8];
  #pragma unroll
  for (int j = 0; j < 2; ++j)
    #pragma unroll
    for (int r = 0; r < 8; ++r) {
      int m = (wave * 2 + j) * 16 + 8 * half + r;   // matches C/D VGPR layout
      x[j][r]  = x0[m * B_ + ng];
      bm[j][r] = B_m1[m];
      ig[j][r] = I_go[m];
      wo[j][r] = Wout[m];
    }
  const int trig  = triggers[ng];
  const int mbase = wave * 32 + 8 * half;           // first global row, j=0

  // initial rm1 image = softplus(x_init)
  #pragma unroll
  for (int j = 0; j < 2; ++j)
    #pragma unroll
    for (int r = 0; r < 8; ++r)
      ldsB[((mbase + j * 16 + r) >> 5) * 512 + (nloc + 16 * (((mbase + j * 16 + r) >> 3) & 1)) * 16 + ((mbase + j * 16 + r) & 7) + (((mbase + j * 16 + r) >> 1) & 8)] = (__bf16)softplus_f(x[j][r]);

  // prefetch noise[t=0]: thread tid fetches row m=tid (16 floats, 4 x b128)
  #pragma unroll
  for (int i = 0; i < 4; ++i)
    async_b128(noise + ((size_t)tid * B_ + col0 + i * 4),
               lds_offset(&ldsN[tid * 16 + i * 4]));

  // Per-lane fragment base pointers (K-block stride = 1024 B = 512 elements)
  const __bf16* ap0 = Jp + ((size_t)rt0 * 32 * 32 + lane) * 16;
  const __bf16* ap1 = Jp + ((size_t)rt1 * 32 * 32 + lane) * 16;
  const __bf16* bp  = &ldsB[lane * 16];

  float loss = 0.0f;
  __syncthreads();

  for (int t = 0; t < T_; ++t) {
    // ---- GEMM phase: acc = J_eff @ rm1 for this wave's two 16-row tiles ----
    // Rolled loop (unroll 2): A fragments stream from L2-resident Jp each
    // step; keeps VGPR pressure low so nothing spills to scratch.
    v8f acc0 = {}, acc1 = {};
    #pragma unroll 2
    for (int tk = 0; tk < 32; ++tk) {
      v16bf b  = *(const v16bf*)(bp  + tk * 512);
      v16bf a0 = *(const v16bf*)(ap0 + tk * 512);
      v16bf a1 = *(const v16bf*)(ap1 + tk * 512);
      acc0 = __builtin_amdgcn_wmma_f32_16x16x32_bf16(false, a0, false, b,
                                                     (short)0, acc0, false, false);
      acc1 = __builtin_amdgcn_wmma_f32_16x16x32_bf16(false, a1, false, b,
                                                     (short)0, acc1, false, false);
    }
    wait_async0();        // this thread's noise[t] async copies are done
    __syncthreads();      // (1) all B reads done, whole noise tile resident

    if (tid < 16) pos_buf[tid] = 0.0f;
    const float go = pulses[t * P_ + trig];
    float partial = 0.0f;
    #pragma unroll
    for (int j = 0; j < 2; ++j)
      #pragma unroll
      for (int r = 0; r < 8; ++r) {
        int   mg   = mbase + j * 16 + r;
        float accv = (j == 0) ? acc0[r] : acc1[r];
        float nz   = ldsN[mg * 16 + nloc];
        float xv   = x[j][r];
        xv += A_COEF * (-xv + accv + bm[j][r] + ig[j][r] * go + nz * NSCALE);
        x[j][r] = xv;
        float s = softplus_f(xv);
        ldsB[(mg >> 5) * 512 + (nloc + 16 * ((mg >> 3) & 1)) * 16 + (mg & 7) + ((mg >> 1) & 8)] = (__bf16)s;           // next step's B operand
        partial = fmaf(wo[j][r], s, partial);       // Wout @ softplus(x_new)
      }
    partial += __shfl_xor(partial, 16);             // fold lane halves (same n)
    __syncthreads();      // (2) noise reads + rm1 writes complete

    if (t + 1 < T_) {     // prefetch next step's noise while we reduce/loop
      #pragma unroll
      for (int i = 0; i < 4; ++i)
        async_b128(noise + ((size_t)(t + 1) * N_ * B_ + (size_t)tid * B_ + col0 + i * 4),
                   lds_offset(&ldsN[tid * 16 + i * 4]));
    }
    if (half == 0) atomicAdd(&pos_buf[nloc], partial);
    __syncthreads();      // (3) position reduction complete

    if (tid < 16) {
      float d = targets[t * B_ + col0 + tid] - pos_buf[tid];
      loss = fmaf(d, d, loss);
    }
  }

  if (tid < 16)
    atomicAdd(out, loss * (1.0f / ((float)T_ * (float)B_)));
}

extern "C" void kernel_launch(void* const* d_in, const int* in_sizes, int n_in,
                              void* d_out, int out_size, void* d_ws, size_t ws_size,
                              hipStream_t stream) {
  const float* targets = (const float*)d_in[0];
  const float* pulses  = (const float*)d_in[1];
  const float* J       = (const float*)d_in[2];
  const float* U       = (const float*)d_in[3];
  const float* V       = (const float*)d_in[4];
  const float* B_m1    = (const float*)d_in[5];
  const float* B_bg    = (const float*)d_in[6];
  const float* Wout    = (const float*)d_in[7];
  const float* I_go    = (const float*)d_in[8];
  const float* x0      = (const float*)d_in[9];
  const float* noise   = (const float*)d_in[10];
  const int*   trig    = (const int*)d_in[11];

  __bf16* Jp   = (__bf16*)d_ws;               // 2 MB packed bf16 J_eff
  float*  outf = (float*)d_out;

  (void)hipMemsetAsync(d_out, 0, sizeof(float), stream);
  jpack_kernel<<<(N_ * N_) / 256, 256, 0, stream>>>(J, U, V, B_bg, Jp);
  net_kernel<<<B_ / 16, 1024, 0, stream>>>(targets, pulses, B_m1, Wout, I_go,
                                           x0, noise, trig, Jp, outf);
}
